// PSWBlockV17_47648367182293
// MI455X (gfx1250) — matrix operand order repaired and checked by reading, not compiled, not run on hardware
//
#include <hip/hip_runtime.h>
#include <hip/hip_bf16.h>
#include <math.h>

#define Bsz   8
#define Lsz   1024
#define Dsz   256
#define Hn    8
#define HDsz  32
#define INNERsz 1024
#define Mrows (Bsz*Lsz)   // 8192

typedef __attribute__((ext_vector_type(16))) __bf16 v16bf;
typedef __attribute__((ext_vector_type(8)))  __bf16 v8bf;
typedef __attribute__((ext_vector_type(8)))  float  v8f;
typedef __attribute__((ext_vector_type(4)))  unsigned uint4v;
typedef __attribute__((ext_vector_type(8)))  int      int8v;
typedef __attribute__((ext_vector_type(4)))  int      int4v;

union BFrag { v16bf v; v8bf h[2]; };

static __device__ __forceinline__ __bf16 f2bf(float f) {
    union { float f; unsigned u; } x; x.f = f;
    unsigned r = (x.u + 0x7fffu + ((x.u >> 16) & 1u)) >> 16;
    union { unsigned short s; __bf16 b; } y; y.s = (unsigned short)r;
    return y.b;
}

// Issue a 2-D TDM tile load (bf16 elements): tile_d0 (contiguous) x tile_d1 rows,
// row stride = stride_el elements, into LDS at byte offset lds_off.
// Descriptor layout per CDNA5 ISA.
// This toolchain exposes the 6-arg builtin (extra int32x8 group before cpol).
static __device__ __forceinline__ void tdm_load_2d(const void* gaddr, unsigned lds_off,
                                                   unsigned tile_d0, unsigned tile_d1,
                                                   unsigned tensor_d0, unsigned tensor_d1,
                                                   unsigned stride_el) {
    unsigned long long ga = (unsigned long long)gaddr;
    uint4v g0;
    g0[0] = 1u;                                        // count=1 user descriptor
    g0[1] = lds_off;                                   // lds_addr (bytes)
    g0[2] = (unsigned)(ga & 0xffffffffu);              // global_addr[31:0]
    g0[3] = (unsigned)((ga >> 32) & 0x1ffffffu) | (2u << 30);  // addr[56:32] | type=2
    int8v g1;
    g1[0] = (int)(1u << 16);                           // data_size=1 (2 bytes)
    g1[1] = (int)(tensor_d0 << 16);                    // tensor_dim0[15:0] @bit48
    g1[2] = (int)(tensor_d1 << 16);                    // tensor_dim1[15:0] @bit80
    g1[3] = (int)(tile_d0 << 16);                      // tile_dim0 @bit112
    g1[4] = (int)(tile_d1 & 0xffffu);                  // tile_dim1 (tile_dim2=0)
    g1[5] = (int)stride_el;                            // tensor_dim0_stride[31:0]
    g1[6] = 0;                                         // stride[47:32] | dim1_stride lo
    g1[7] = 0;
    int4v gz4 = {0, 0, 0, 0};
    int8v gz8 = {0, 0, 0, 0, 0, 0, 0, 0};
    __builtin_amdgcn_tensor_load_to_lds(g0, g1, gz4, gz4, gz8, 0);
}

// ---------------- conversion / precompute kernels ----------------

__global__ void convert_bf16_kernel(const float* __restrict__ src,
                                    __bf16* __restrict__ dst, int n) {
    int i = blockIdx.x * blockDim.x + threadIdx.x;
    if (i < n) dst[i] = f2bf(src[i]);
}

__global__ void cmsm_kernel(const float* __restrict__ phi,
                            const float* __restrict__ mag,
                            float* __restrict__ cm, float* __restrict__ sm, int n) {
    int i = blockIdx.x * blockDim.x + threadIdx.x;
    if (i < n) {
        float p = phi[i], m = mag[i];
        cm[i] = m * __cosf(p);
        sm[i] = m * __sinf(p);
    }
}

// ---------------- bf16 WMMA GEMM with TDM double-buffered LDS staging ----------------
//  C[M,N] = A[M,K] * W[N,K]^T + bias
// block = 4 waves, 64x64 C tile; per k-step TDM loads A(64x32) and B(64x32) into LDS.
// modes: 1 = bf16 store to [B,H,L,HD] (scaled)   2 = bf16 store to [B,H,HD,L] (V transposed)
//        3 = fp32 store row-major [M,N]          4 = erf-GELU then bf16 store row-major

#define TILE_ELEMS (64 * 32)

__global__ __launch_bounds__(128)
void gemm_wmma_kernel(const __bf16* __restrict__ A, const __bf16* __restrict__ W,
                      const float* __restrict__ bias, void* __restrict__ out,
                      int M, int N, int K, int mode, float scale) {
    __shared__ __bf16 sA[2][TILE_ELEMS];
    __shared__ __bf16 sB[2][TILE_ELEMS];

    const int wave = threadIdx.x >> 5;
    const int lane = threadIdx.x & 31;
    const int lm   = lane & 15;
    const int hi   = (lane >> 4) & 1;

    const int row0 = blockIdx.y * 64 + wave * 16;   // this wave's 16 rows
    const int col0 = blockIdx.x * 64;

    const __bf16* Ablk = A + (size_t)(blockIdx.y * 64) * K;  // 64-row strip
    const __bf16* Wblk = W + (size_t)col0 * K;               // 64-col strip

    const int ksteps = K >> 5;
    const bool issuer = (threadIdx.x < 32);   // wave 0 drives the TDM

    if (issuer) {
        tdm_load_2d(Ablk, (unsigned)(size_t)(void*)&sA[0][0], 32, 64, (unsigned)K, 64, (unsigned)K);
        tdm_load_2d(Wblk, (unsigned)(size_t)(void*)&sB[0][0], 32, 64, (unsigned)K, 64, (unsigned)K);
    }

    v8f acc[4] = {};
    for (int ks = 0; ks < ksteps; ++ks) {
        const int cur = ks & 1;
        if (issuer) {
            if (ks + 1 < ksteps) {
                const int nxt = cur ^ 1;
                tdm_load_2d(Ablk + (ks + 1) * 32, (unsigned)(size_t)(void*)&sA[nxt][0],
                            32, 64, (unsigned)K, 64, (unsigned)K);
                tdm_load_2d(Wblk + (ks + 1) * 32, (unsigned)(size_t)(void*)&sB[nxt][0],
                            32, 64, (unsigned)K, 64, (unsigned)K);
                __builtin_amdgcn_s_wait_tensorcnt(2);  // current pair complete, next in flight
            } else {
                __builtin_amdgcn_s_wait_tensorcnt(0);
            }
        }
        __syncthreads();   // publish LDS tile to all 4 waves

        BFrag fa;
        const __bf16* ap = &sA[cur][(wave * 16 + lm) * 32 + (hi ? 8 : 0)];
        fa.h[0] = *(const v8bf*)(ap);
        fa.h[1] = *(const v8bf*)(ap + 16);
#pragma unroll
        for (int t = 0; t < 4; ++t) {
            BFrag fb;
            const __bf16* bp = &sB[cur][(t * 16 + lm) * 32 + (hi ? 8 : 0)];
            fb.h[0] = *(const v8bf*)(bp);
            fb.h[1] = *(const v8bf*)(bp + 16);
            acc[t] = __builtin_amdgcn_wmma_f32_16x16x32_bf16(
                false, fa.v, false, fb.v, (short)0, acc[t], false, false);
        }
        __syncthreads();   // done reading before this buffer is re-filled
    }

#pragma unroll
    for (int t = 0; t < 4; ++t) {
#pragma unroll
        for (int r = 0; r < 8; ++r) {
            int m = row0 + r + hi * 8;
            int n = col0 + t * 16 + lm;
            float v = acc[t][r] + bias[n];
            if (mode == 1) {
                int b = m / Lsz, l = m % Lsz, hh = n / HDsz, hd = n % HDsz;
                ((__bf16*)out)[(((size_t)(b * Hn + hh)) * Lsz + l) * HDsz + hd] = f2bf(v * scale);
            } else if (mode == 2) {
                int b = m / Lsz, l = m % Lsz, hh = n / HDsz, hd = n % HDsz;
                ((__bf16*)out)[(((size_t)(b * Hn + hh)) * HDsz + hd) * Lsz + l] = f2bf(v);
            } else if (mode == 3) {
                ((float*)out)[(size_t)m * N + n] = v;
            } else { // 4: exact-erf GELU -> bf16
                float g = 0.5f * v * (1.0f + erff(v * 0.70710678118654752f));
                ((__bf16*)out)[(size_t)m * N + n] = f2bf(g);
            }
        }
    }
}

// ---------------- attention: 1 wave per (b,h,16-query tile) ----------------
// LDS: fp32 scores [16][1024] + bf16 P [16][1024] + row sums [16]

__global__ __launch_bounds__(32)
void attn_kernel(const __bf16* __restrict__ qbh, const __bf16* __restrict__ kbh,
                 const __bf16* __restrict__ vT, const float* __restrict__ mask,
                 const float* __restrict__ cm, const float* __restrict__ sm,
                 const float* __restrict__ gamma, __bf16* __restrict__ ctx) {
    extern __shared__ char smem[];
    float*  sS   = (float*)smem;                          // 16*1024 fp32
    __bf16* sP   = (__bf16*)(smem + 16 * 1024 * 4);       // 16*1024 bf16
    float*  sSum = (float*)(smem + 16 * 1024 * 4 + 16 * 1024 * 2);

    const int lane = threadIdx.x;
    const int lm   = lane & 15;
    const int hi   = (lane >> 4) & 1;

    const int qt = blockIdx.x & 63;          // L/16 = 64
    const int h  = (blockIdx.x >> 6) & 7;
    const int b  = blockIdx.x >> 9;
    const size_t bh = (size_t)(b * Hn + h);

    // Q A-fragment (16 rows x K=HD=32), reused across all key tiles
    BFrag fq;
    {
        const __bf16* qp = qbh + (bh * Lsz + qt * 16 + lm) * HDsz + (hi ? 8 : 0);
        fq.h[0] = *(const v8bf*)(qp);
        fq.h[1] = *(const v8bf*)(qp + 16);
    }
    float cmq[8], smq[8];
#pragma unroll
    for (int r = 0; r < 8; ++r) {
        int m = qt * 16 + r + hi * 8;
        cmq[r] = cm[bh * Lsz + m];
        smq[r] = sm[bh * Lsz + m];
    }
    const float gam = gamma[h];

    // ---- scores: 64 key tiles, one WMMA each (K-dim = HD = 32) ----
    for (int kt = 0; kt < 64; ++kt) {
        BFrag fk;
        const __bf16* kp = kbh + (bh * Lsz + kt * 16 + lm) * HDsz + (hi ? 8 : 0);
        fk.h[0] = *(const v8bf*)(kp);
        fk.h[1] = *(const v8bf*)(kp + 16);
        v8f c = {};
        c = __builtin_amdgcn_wmma_f32_16x16x32_bf16(false, fq.v, false, fk.v,
                                                    (short)0, c, false, false);
        int n = kt * 16 + lm;
        float am  = mask[(size_t)b * Lsz + n];
        float cmk = cm[bh * Lsz + n];
        float smk = sm[bh * Lsz + n];
#pragma unroll
        for (int r = 0; r < 8; ++r) {
            float s = c[r] + am + gam * (cmq[r] * cmk + smq[r] * smk);
            sS[(size_t)(r + hi * 8) * Lsz + n] = s;
        }
    }

    // ---- softmax per row (wave-wide shfl reductions) ----
    for (int r2 = 0; r2 < 16; ++r2) {
        const float* row = sS + (size_t)r2 * Lsz;
        float mx = -1e30f;
        for (int j = lane; j < Lsz; j += 32) mx = fmaxf(mx, row[j]);
#pragma unroll
        for (int off = 16; off > 0; off >>= 1) mx = fmaxf(mx, __shfl_xor(mx, off, 32));
        const int qglob = qt * 16 + r2;
        const bool deg = (mx <= -1e8f);   // fully-masked row -> one-hot at diagonal
        float sum = 0.f;
        for (int j = lane; j < Lsz; j += 32) {
            float p = deg ? ((j == qglob) ? 1.f : 0.f) : __expf(row[j] - mx);
            sum += p;
            sP[(size_t)r2 * Lsz + j] = f2bf(p);
        }
#pragma unroll
        for (int off = 16; off > 0; off >>= 1) sum += __shfl_xor(sum, off, 32);
        if (lane == 0) sSum[r2] = sum;
    }

    // ---- P @ V : 32 key chunks x 2 d-tiles ----
    v8f o0 = {}, o1 = {};
    const __bf16* vrow0 = vT + (bh * HDsz + 0  + lm) * Lsz;
    const __bf16* vrow1 = vT + (bh * HDsz + 16 + lm) * Lsz;
    for (int kk = 0; kk < 32; ++kk) {
        BFrag fa;
        const __bf16* pp = sP + (size_t)lm * Lsz + kk * 32 + (hi ? 8 : 0);
        fa.h[0] = *(const v8bf*)(pp);
        fa.h[1] = *(const v8bf*)(pp + 16);
        int koff = kk * 32 + (hi ? 8 : 0);
        BFrag fb0, fb1;
        fb0.h[0] = *(const v8bf*)(vrow0 + koff);
        fb0.h[1] = *(const v8bf*)(vrow0 + koff + 16);
        fb1.h[0] = *(const v8bf*)(vrow1 + koff);
        fb1.h[1] = *(const v8bf*)(vrow1 + koff + 16);
        o0 = __builtin_amdgcn_wmma_f32_16x16x32_bf16(false, fa.v, false, fb0.v,
                                                     (short)0, o0, false, false);
        o1 = __builtin_amdgcn_wmma_f32_16x16x32_bf16(false, fa.v, false, fb1.v,
                                                     (short)0, o1, false, false);
    }

#pragma unroll
    for (int r = 0; r < 8; ++r) {
        int m = r + hi * 8;
        float inv = 1.f / sSum[m];
        size_t tok = (size_t)b * Lsz + qt * 16 + m;
        ctx[tok * Dsz + h * HDsz + lm]      = f2bf(o0[r] * inv);
        ctx[tok * Dsz + h * HDsz + 16 + lm] = f2bf(o1[r] * inv);
    }
}

// ---------------- residual + LayerNorm (one block per row, D=256) ----------------

__global__ __launch_bounds__(256)
void ln_kernel(const float* __restrict__ pre, const float* __restrict__ resid,
               const float* __restrict__ g, const float* __restrict__ bta,
               float* __restrict__ outf, __bf16* __restrict__ outb) {
    __shared__ float red[256];
    const int row = blockIdx.x, t = threadIdx.x;
    const size_t idx = (size_t)row * Dsz + t;
    float x = pre[idx] + resid[idx];
    red[t] = x; __syncthreads();
#pragma unroll
    for (int s = 128; s > 0; s >>= 1) { if (t < s) red[t] += red[t + s]; __syncthreads(); }
    float mean = red[0] * (1.0f / Dsz); __syncthreads();
    float d = x - mean;
    red[t] = d * d; __syncthreads();
#pragma unroll
    for (int s = 128; s > 0; s >>= 1) { if (t < s) red[t] += red[t + s]; __syncthreads(); }
    float var = red[0] * (1.0f / Dsz);
    float y = d * rsqrtf(var + 1e-12f) * g[t] + bta[t];
    outf[idx] = y;
    if (outb) outb[idx] = f2bf(y);
}

// ---------------- host launch ----------------

extern "C" void kernel_launch(void* const* d_in, const int* in_sizes, int n_in,
                              void* d_out, int out_size, void* d_ws, size_t ws_size,
                              hipStream_t stream) {
    const float* hs   = (const float*)d_in[0];
    const float* mask = (const float*)d_in[1];
    const float* phi  = (const float*)d_in[2];
    const float* mag  = (const float*)d_in[3];
    const float* Wq = (const float*)d_in[4];  const float* bq = (const float*)d_in[5];
    const float* Wk = (const float*)d_in[6];  const float* bk = (const float*)d_in[7];
    const float* Wv = (const float*)d_in[8];  const float* bv = (const float*)d_in[9];
    const float* Wo = (const float*)d_in[10]; const float* bo = (const float*)d_in[11];
    const float* ln1g = (const float*)d_in[12]; const float* ln1b = (const float*)d_in[13];
    const float* gamma = (const float*)d_in[14];
    const float* W1 = (const float*)d_in[15]; const float* b1 = (const float*)d_in[16];
    const float* W2 = (const float*)d_in[17]; const float* b2 = (const float*)d_in[18];
    const float* ln2g = (const float*)d_in[19]; const float* ln2b = (const float*)d_in[20];
    float* out = (float*)d_out;

    // workspace carving (256B aligned)
    char* ws = (char*)d_ws;
    size_t off = 0;
    auto carve = [&](size_t bytes) { void* p = ws + off; off = (off + bytes + 255) & ~(size_t)255; return p; };
    __bf16* hs_bf   = (__bf16*)carve((size_t)Mrows * Dsz * 2);
    __bf16* Wq_bf   = (__bf16*)carve((size_t)Dsz * Dsz * 2);
    __bf16* Wk_bf   = (__bf16*)carve((size_t)Dsz * Dsz * 2);
    __bf16* Wv_bf   = (__bf16*)carve((size_t)Dsz * Dsz * 2);
    __bf16* Wo_bf   = (__bf16*)carve((size_t)Dsz * Dsz * 2);
    __bf16* W1_bf   = (__bf16*)carve((size_t)INNERsz * Dsz * 2);
    __bf16* W2_bf   = (__bf16*)carve((size_t)Dsz * INNERsz * 2);
    float*  cm      = (float*)carve((size_t)Bsz * Hn * Lsz * 4);
    float*  sm      = (float*)carve((size_t)Bsz * Hn * Lsz * 4);
    __bf16* qbh     = (__bf16*)carve((size_t)Mrows * Dsz * 2);
    __bf16* kbh     = (__bf16*)carve((size_t)Mrows * Dsz * 2);
    __bf16* vT      = (__bf16*)carve((size_t)Mrows * Dsz * 2);
    __bf16* ctx_bf  = (__bf16*)carve((size_t)Mrows * Dsz * 2);
    float*  preO    = (float*)carve((size_t)Mrows * Dsz * 4);   // reused for pre2
    float*  attn    = (float*)carve((size_t)Mrows * Dsz * 4);
    __bf16* attn_bf = (__bf16*)carve((size_t)Mrows * Dsz * 2);
    __bf16* h1      = (__bf16*)carve((size_t)Mrows * INNERsz * 2);

    auto conv = [&](const float* s, __bf16* d, int n) {
        convert_bf16_kernel<<<(n + 255) / 256, 256, 0, stream>>>(s, d, n);
    };
    conv(hs, hs_bf, Mrows * Dsz);
    conv(Wq, Wq_bf, Dsz * Dsz);   conv(Wk, Wk_bf, Dsz * Dsz);
    conv(Wv, Wv_bf, Dsz * Dsz);   conv(Wo, Wo_bf, Dsz * Dsz);
    conv(W1, W1_bf, INNERsz * Dsz); conv(W2, W2_bf, Dsz * INNERsz);

    cmsm_kernel<<<(Bsz * Hn * Lsz + 255) / 256, 256, 0, stream>>>(phi, mag, cm, sm, Bsz * Hn * Lsz);

    const float qscale = 0.17677669529663687f;  // 1/sqrt(HD)
    dim3 blk(128);
    // Q/K/V projections
    gemm_wmma_kernel<<<dim3(Dsz / 64, Mrows / 64), blk, 0, stream>>>(
        hs_bf, Wq_bf, bq, qbh, Mrows, Dsz, Dsz, 1, qscale);
    gemm_wmma_kernel<<<dim3(Dsz / 64, Mrows / 64), blk, 0, stream>>>(
        hs_bf, Wk_bf, bk, kbh, Mrows, Dsz, Dsz, 1, 1.0f);
    gemm_wmma_kernel<<<dim3(Dsz / 64, Mrows / 64), blk, 0, stream>>>(
        hs_bf, Wv_bf, bv, vT, Mrows, Dsz, Dsz, 2, 1.0f);

    // attention (physics bias as rank-2 cm/sm term, LDS-resident scores)
    size_t attn_lds = (size_t)16 * Lsz * 4 + (size_t)16 * Lsz * 2 + 64 * 4;
    attn_kernel<<<Bsz * Hn * (Lsz / 16), 32, attn_lds, stream>>>(
        qbh, kbh, vT, mask, cm, sm, gamma, ctx_bf);

    // O projection -> fp32, then residual+LN1
    gemm_wmma_kernel<<<dim3(Dsz / 64, Mrows / 64), blk, 0, stream>>>(
        ctx_bf, Wo_bf, bo, preO, Mrows, Dsz, Dsz, 3, 1.0f);
    ln_kernel<<<Mrows, 256, 0, stream>>>(preO, hs, ln1g, ln1b, attn, attn_bf);

    // MLP: GEMM+GELU -> bf16, GEMM -> fp32, residual+LN2 -> out
    gemm_wmma_kernel<<<dim3(INNERsz / 64, Mrows / 64), blk, 0, stream>>>(
        attn_bf, W1_bf, b1, h1, Mrows, INNERsz, Dsz, 4, 1.0f);
    gemm_wmma_kernel<<<dim3(Dsz / 64, Mrows / 64), blk, 0, stream>>>(
        h1, W2_bf, b2, preO, Mrows, Dsz, INNERsz, 3, 1.0f);
    ln_kernel<<<Mrows, 256, 0, stream>>>(preO, attn, ln2g, ln2b, out, (__bf16*)nullptr);
}
